// RNNNERModel_51505247814050
// MI455X (gfx1250) — matrix operand order repaired and checked by reading, not compiled, not run on hardware
//
#include <hip/hip_runtime.h>

// ---------------------------------------------------------------------------
// BiLSTM(2 layers, H=512) + FC(C=9) + CRF loss for gfx1250 (MI455X).
// v_wmma_f32_16x16x32_bf16 with f32 accumulate; 32x32 per-wave GEMM tiles;
// persistent recurrence kernel with device-scope software barrier, h_prev
// staged to LDS via global_load_async_to_lds_b128 (ASYNCcnt path), xg
// prefetched one step ahead, cell state register-resident across 512 steps.
// ---------------------------------------------------------------------------

typedef __bf16 bf16_t;
typedef __attribute__((ext_vector_type(16))) __bf16 v16bf;
typedef __attribute__((ext_vector_type(8)))  __bf16 v8bf;
typedef __attribute__((ext_vector_type(8)))  float  v8f;
typedef __attribute__((__vector_size__(16))) int    v4i;

#define Bz 32
#define Lz 512
#define Ez 256
#define Hz 512
#define Cz 9
#define Mz (Bz * Lz)   // 16384 rows
#define G4H 2048       // 4*H gate width
#define SCAN_WGS 16

#if defined(__HIP_DEVICE_COMPILE__) && __has_builtin(__builtin_amdgcn_global_load_async_to_lds_b128)
#define HAVE_ASYNC_LDS 1
#endif

__device__ __forceinline__ float sigmoidf_(float x) { return 1.0f / (1.0f + __expf(-x)); }

// ---- device-scope software barrier (sense via monotone generation) --------
__device__ __forceinline__ void grid_barrier(unsigned* sync, unsigned& bar_t, unsigned nwg) {
  __threadfence();          // each thread flushes its own global stores
  __syncthreads();
  if (threadIdx.x == 0) {
    unsigned arr = atomicAdd(&sync[0], 1u);
    if (arr == nwg - 1u) {
      atomicExch(&sync[0], 0u);
      __threadfence();
      atomicAdd(&sync[1], 1u);
    }
    ++bar_t;
    volatile unsigned* gv = &sync[1];
    while ((int)(*gv - bar_t) < 0) __builtin_amdgcn_s_sleep(1);
  }
  __syncthreads();
}

__global__ void zero_sync_kernel(unsigned* s) {
  if (threadIdx.x < 16) s[threadIdx.x] = 0u;
}

// ---- embedding gather: xbf[m, 0:256] = bf16(emb[X[m], :]) -----------------
__global__ void embed_gather(const int* __restrict__ X, const float* __restrict__ emb,
                             bf16_t* __restrict__ xbf) {
  int row = blockIdx.x;          // m = b*L + l
  int e   = threadIdx.x;         // 0..255
  int tok = X[row];
  xbf[(size_t)row * Ez + e] = (bf16_t)emb[(size_t)tok * Ez + e];
}

// ---- f32 -> bf16 convert ---------------------------------------------------
__global__ void f32_to_bf16(const float* __restrict__ s, bf16_t* __restrict__ d, int n) {
  int i = blockIdx.x * blockDim.x + threadIdx.x;
  if (i < n) d[i] = (bf16_t)s[i];
}

// ---- input projection GEMM: Out(M,2048) = A(M,K)bf16 @ W(2048,K)^T + b ----
// 256 threads = 8 waves; WG tile 64(M) x 128(N); 32x32 per wave
// (2 A frags x 2 B frags -> 4 WMMA per 8 b128 loads).
__global__ void gemm_xg(const bf16_t* __restrict__ A, const bf16_t* __restrict__ Wbf,
                        const float* __restrict__ b_ih, const float* __restrict__ b_hh,
                        float* __restrict__ Out, int K) {
  const int lane = threadIdx.x & 31;
  const int wv   = threadIdx.x >> 5;
  const int half = lane >> 4;
  const int l15  = lane & 15;
  const int m0 = blockIdx.y * 64 + (wv >> 2) * 32;
  const int n0 = blockIdx.x * 128 + (wv & 3) * 32;

  const v8f vz = {0.f, 0.f, 0.f, 0.f, 0.f, 0.f, 0.f, 0.f};
  v8f acc[2][2];
#pragma unroll
  for (int i = 0; i < 2; ++i)
#pragma unroll
    for (int j = 0; j < 2; ++j) acc[i][j] = vz;

  const bf16_t* ar[2] = {A + (size_t)(m0 + l15) * K, A + (size_t)(m0 + 16 + l15) * K};
  const bf16_t* wr[2] = {Wbf + (size_t)(n0 + l15) * K, Wbf + (size_t)(n0 + 16 + l15) * K};
  for (int k0 = 0; k0 < K; k0 += 32) {
    union U { v16bf v; v8bf h[2]; } a[2];
    v16bf b[2];
#pragma unroll
    for (int i = 0; i < 2; ++i) {
      a[i].h[0] = *(const v8bf*)(ar[i] + k0 + half * 16);      // K = k0+half*16+0..7
      a[i].h[1] = *(const v8bf*)(ar[i] + k0 + 8 + half * 16);  // K = k0+half*16+8..15
    }
#pragma unroll
    for (int j = 0; j < 2; ++j)
      b[j] = *(const v16bf*)(wr[j] + k0 + half * 16);          // K = k0+half*16+0..15
#pragma unroll
    for (int i = 0; i < 2; ++i)
#pragma unroll
      for (int j = 0; j < 2; ++j)
        acc[i][j] = __builtin_amdgcn_wmma_f32_16x16x32_bf16(false, a[i].v, false, b[j],
                                                            (short)0, acc[i][j], false, false);
  }
#pragma unroll
  for (int j = 0; j < 2; ++j) {
    const int n = n0 + j * 16 + l15;
    const float bias = b_ih[n] + b_hh[n];
#pragma unroll
    for (int i = 0; i < 2; ++i)
#pragma unroll
      for (int r = 0; r < 8; ++r) {
        int mr = m0 + i * 16 + r + half * 8;                   // D: VGPR r -> M=r+8*half
        Out[(size_t)mr * G4H + n] = acc[i][j][r] + bias;
      }
  }
}

// ---- persistent LSTM scan: 16 WGs, each owns a 32-column h slice ----------
__global__ void lstm_scan(const float* __restrict__ xg, const bf16_t* __restrict__ whh,
                          float* __restrict__ hcat, bf16_t* __restrict__ hbuf,
                          const int reverse, const int dirofs, unsigned* sync) {
  __shared__ bf16_t hs[Bz * Hz];   // staged h_prev (32 KB)
  __shared__ float  gl[Bz * 128];  // 32 batches x (4 gates x 32 cols) slice (16 KB)
  const int tid  = threadIdx.x;
  const int wg   = blockIdx.x;
  const int lane = tid & 31, wv = tid >> 5;
  const int half = lane >> 4, l15 = lane & 15;
  const int mt = wv >> 2;          // 0..1 : which 16 batches
  const int q  = wv & 3;           // gate quadrant: 0=i 1=f 2=g 3=o

  // zero h double-buffer (h_prev for step 0 is zero)
  for (int i = tid + wg * 256; i < 2 * Bz * Hz; i += SCAN_WGS * 256)
    hbuf[i] = (bf16_t)0.0f;

  float creg[4] = {0.f, 0.f, 0.f, 0.f};  // cell state lives in registers

  unsigned bar_t = 0;
  if (tid == 0) bar_t = *(volatile unsigned*)(sync + 1);
  grid_barrier(sync, bar_t, gridDim.x);

  const int n0g = q * Hz + wg * 32 + l15;       // this wave's first gate column
  const int n1g = n0g + 16;
  const bf16_t* w0 = whh + (size_t)n0g * Hz;
  const bf16_t* w1 = whh + (size_t)n1g * Hz;
  const int arow_l = (mt * 16 + l15) * Hz;      // A row offset inside hs (elements)

  for (int t = 0; t < Lz; ++t) {
    const int tt = reverse ? (Lz - 1 - t) : t;
    const int ttn = reverse ? (tt - 1) : (tt + 1);
    const bf16_t* hread  = hbuf + ((t & 1) ? Bz * Hz : 0);
    bf16_t*       hwrite = hbuf + ((t & 1) ? 0 : Bz * Hz);

    // ---- stage h_prev (32 KB) into LDS -----------------------------------
#ifdef HAVE_ASYNC_LDS
    {
      char* gsrc = (char*)hread;
      char* ldst = (char*)hs;
      for (int i = tid; i < 2048; i += 256)
        __builtin_amdgcn_global_load_async_to_lds_b128(
            (v4i*)(gsrc + (size_t)i * 16),
            (__attribute__((address_space(3))) v4i*)(ldst + (size_t)i * 16), 0, 0);
#if __has_builtin(__builtin_amdgcn_s_wait_asynccnt)
      __builtin_amdgcn_s_wait_asynccnt(0);
#else
      asm volatile("s_wait_asynccnt 0x0" ::: "memory");
#endif
    }
#else
    {
      ulonglong2* d = (ulonglong2*)hs;
      const ulonglong2* s2 = (const ulonglong2*)hread;
      for (int i = tid; i < 2048; i += 256) d[i] = s2[i];
    }
#endif
    __syncthreads();

    // ---- gates(32 x 128 slice) = h_prev @ whh^T slice + xg[:, tt, slice] --
    v8f acc0, acc1;
#pragma unroll
    for (int r = 0; r < 8; ++r) {
      const int b = mt * 16 + r + half * 8;
      const size_t rb = ((size_t)b * Lz + tt) * G4H;
      acc0[r] = xg[rb + n0g];
      acc1[r] = xg[rb + n1g];
      if (t + 1 < Lz) {  // prefetch next step's xg slice (global_prefetch_b8)
        const size_t rn = ((size_t)b * Lz + ttn) * G4H;
        __builtin_prefetch(&xg[rn + n0g], 0, 1);
      }
    }
    for (int k0 = 0; k0 < Hz; k0 += 32) {
      union U { v16bf v; v8bf h[2]; } au;                 // A from LDS (ds_load_b128)
      au.h[0] = *(const v8bf*)(hs + arow_l + k0 + half * 16);
      au.h[1] = *(const v8bf*)(hs + arow_l + k0 + 8 + half * 16);
      v16bf b0 = *(const v16bf*)(w0 + k0 + half * 16);
      v16bf b1 = *(const v16bf*)(w1 + k0 + half * 16);
      acc0 = __builtin_amdgcn_wmma_f32_16x16x32_bf16(false, au.v, false, b0,
                                                     (short)0, acc0, false, false);
      acc1 = __builtin_amdgcn_wmma_f32_16x16x32_bf16(false, au.v, false, b1,
                                                     (short)0, acc1, false, false);
    }
#pragma unroll
    for (int r = 0; r < 8; ++r) {
      const int b = mt * 16 + r + half * 8;
      gl[b * 128 + q * 32 + l15]      = acc0[r];
      gl[b * 128 + q * 32 + 16 + l15] = acc1[r];
    }
    __syncthreads();

    // ---- LSTM cell update, WG-local (4 h elems/thread, fixed across steps)
#pragma unroll
    for (int e = 0; e < 4; ++e) {
      int idx = tid * 4 + e;
      int b = idx >> 5, jl = idx & 31;
      float ig = gl[b * 128 + jl];
      float fg = gl[b * 128 + 32 + jl];
      float gg = gl[b * 128 + 64 + jl];
      float og = gl[b * 128 + 96 + jl];
      float c = sigmoidf_(fg) * creg[e] + sigmoidf_(ig) * tanhf(gg);
      float h = sigmoidf_(og) * tanhf(c);
      creg[e] = c;
      int j = wg * 32 + jl;
      hcat[((size_t)b * Lz + tt) * (2 * Hz) + dirofs + j] = h;
      hwrite[b * Hz + j] = (bf16_t)h;
    }
    grid_barrier(sync, bar_t, gridDim.x);  // h_t complete; hs/gl safe to reuse
  }
}

// ---- FC: logits(M,9) = hcat(M,1024) @ fc_w(9,1024)^T + fc_b ---------------
__global__ void fc_logits(const float* __restrict__ hcat, const float* __restrict__ fcw,
                          const float* __restrict__ fcb, float* __restrict__ out) {
  __shared__ float wlds[Cz * 1024];
  __shared__ float psum[8][Cz];
  for (int i = threadIdx.x; i < Cz * 1024; i += 256) wlds[i] = fcw[i];
  __syncthreads();
  int m = blockIdx.x;
  float p[Cz] = {};
  for (int k = threadIdx.x; k < 1024; k += 256) {
    float x = hcat[(size_t)m * 1024 + k];
#pragma unroll
    for (int n = 0; n < Cz; ++n) p[n] += x * wlds[n * 1024 + k];
  }
  int lane = threadIdx.x & 31, wv = threadIdx.x >> 5;
#pragma unroll
  for (int n = 0; n < Cz; ++n) {
    float v = p[n];
    for (int off = 16; off; off >>= 1) v += __shfl_xor(v, off, 32);
    if (lane == 0) psum[wv][n] = v;
  }
  __syncthreads();
  if (threadIdx.x < Cz) {
    float s = fcb[threadIdx.x];
    for (int w = 0; w < 8; ++w) s += psum[w][threadIdx.x];
    out[(size_t)m * Cz + threadIdx.x] = s;
  }
}

// ---- CRF negative log-likelihood (loss = sum(denom - num)) ----------------
__global__ void crf_loss(const float* __restrict__ logits, const int* __restrict__ labels,
                         const float* __restrict__ startv, const float* __restrict__ endv,
                         const float* __restrict__ trans, float* __restrict__ loss_out) {
  __shared__ float tr[Cz * Cz];
  __shared__ float red[Bz];
  if (threadIdx.x < Cz * Cz) tr[threadIdx.x] = trans[threadIdx.x];
  __syncthreads();
  int b = threadIdx.x;
  if (b < Bz) {
    const float* lg = logits + (size_t)b * Lz * Cz;
    const int*   lb = labels + (size_t)b * Lz;
    float alpha[Cz];
    int m0  = lb[0] > -1;
    int tag0 = m0 ? lb[0] : 0;
    float num = startv[tag0] + lg[tag0];
    int cnt = m0 ? 1 : 0;
    int prev = tag0;
#pragma unroll
    for (int c = 0; c < Cz; ++c) alpha[c] = startv[c] + lg[c];
    for (int t = 1; t < Lz; ++t) {
      const float* e = lg + t * Cz;
      int mv = lb[t] > -1;
      int tg = mv ? lb[t] : 0;
      if (mv) { num += tr[prev * Cz + tg] + e[tg]; cnt++; }
      prev = tg;
      if (mv) {
        float na[Cz];
#pragma unroll
        for (int c = 0; c < Cz; ++c) {
          float mx = -1e30f;
#pragma unroll
          for (int cp = 0; cp < Cz; ++cp) mx = fmaxf(mx, alpha[cp] + tr[cp * Cz + c]);
          float s = 0.f;
#pragma unroll
          for (int cp = 0; cp < Cz; ++cp) s += __expf(alpha[cp] + tr[cp * Cz + c] - mx);
          na[c] = mx + __logf(s) + e[c];
        }
#pragma unroll
        for (int c = 0; c < Cz; ++c) alpha[c] = na[c];
      }
    }
    int li = cnt - 1; if (li < 0) li = Lz - 1;  // python -1 wrap
    int lr = lb[li];
    num += endv[lr > -1 ? lr : 0];
    float mx = -1e30f;
#pragma unroll
    for (int c = 0; c < Cz; ++c) mx = fmaxf(mx, alpha[c] + endv[c]);
    float s = 0.f;
#pragma unroll
    for (int c = 0; c < Cz; ++c) s += __expf(alpha[c] + endv[c] - mx);
    red[b] = (mx + __logf(s)) - num;
  }
  __syncthreads();
  if (threadIdx.x == 0) {
    float s = 0.f;
    for (int i = 0; i < Bz; ++i) s += red[i];
    loss_out[0] = s;
  }
}

// ---------------------------------------------------------------------------
extern "C" void kernel_launch(void* const* d_in, const int* in_sizes, int n_in,
                              void* d_out, int out_size, void* d_ws, size_t ws_size,
                              hipStream_t stream) {
  (void)in_sizes; (void)n_in; (void)out_size; (void)ws_size;
  const int*   X      = (const int*)d_in[0];
  const int*   labels = (const int*)d_in[1];
  const float* emb    = (const float*)d_in[2];
  // scans in order: [l0 fwd, l0 rev, l1 fwd, l1 rev]
  const float* wih[4] = {(const float*)d_in[3],  (const float*)d_in[7],
                         (const float*)d_in[11], (const float*)d_in[15]};
  const float* whh[4] = {(const float*)d_in[4],  (const float*)d_in[8],
                         (const float*)d_in[12], (const float*)d_in[16]};
  const float* bih[4] = {(const float*)d_in[5],  (const float*)d_in[9],
                         (const float*)d_in[13], (const float*)d_in[17]};
  const float* bhh[4] = {(const float*)d_in[6],  (const float*)d_in[10],
                         (const float*)d_in[14], (const float*)d_in[18]};
  const float* fcw    = (const float*)d_in[19];
  const float* fcb    = (const float*)d_in[20];
  const float* cstart = (const float*)d_in[21];
  const float* cend   = (const float*)d_in[22];
  const float* ctrans = (const float*)d_in[23];
  float* out = (float*)d_out;   // [0]=loss, [1..]=logits (B,L,C)

  char* base = (char*)d_ws;
  size_t off = 0;
  auto alloc = [&](size_t bytes) -> void* {
    void* p = base + off;
    off = (off + bytes + 255) & ~(size_t)255;
    return p;
  };
  unsigned* sync = (unsigned*)alloc(256);
  bf16_t*   xbf  = (bf16_t*)alloc((size_t)Mz * 1024 * sizeof(bf16_t));   // 32 MB
  float*    xg   = (float*)alloc((size_t)Mz * G4H * sizeof(float));      // 128 MB
  float*    hcat = (float*)alloc((size_t)Mz * 1024 * sizeof(float));     // 64 MB
  bf16_t*   hbuf = (bf16_t*)alloc((size_t)2 * Bz * Hz * sizeof(bf16_t)); // 64 KB
  const int Ks[4] = {Ez, Ez, 2 * Hz, 2 * Hz};
  bf16_t* wihbf[4]; bf16_t* whhbf[4];
  for (int i = 0; i < 4; ++i) wihbf[i] = (bf16_t*)alloc((size_t)G4H * Ks[i] * sizeof(bf16_t));
  for (int i = 0; i < 4; ++i) whhbf[i] = (bf16_t*)alloc((size_t)G4H * Hz * sizeof(bf16_t));

  zero_sync_kernel<<<1, 64, 0, stream>>>(sync);
  embed_gather<<<Mz, Ez, 0, stream>>>(X, emb, xbf);
  for (int i = 0; i < 4; ++i) {
    int n = G4H * Ks[i];
    f32_to_bf16<<<(n + 255) / 256, 256, 0, stream>>>(wih[i], wihbf[i], n);
    int n2 = G4H * Hz;
    f32_to_bf16<<<(n2 + 255) / 256, 256, 0, stream>>>(whh[i], whhbf[i], n2);
  }
  for (int layer = 0; layer < 2; ++layer) {
    if (layer == 1) {
      int n = Mz * 1024;
      f32_to_bf16<<<(n + 255) / 256, 256, 0, stream>>>(hcat, xbf, n);
    }
    for (int dir = 0; dir < 2; ++dir) {
      int s = layer * 2 + dir;
      dim3 g1(G4H / 128, Mz / 64);
      gemm_xg<<<g1, 256, 0, stream>>>(xbf, wihbf[s], bih[s], bhh[s], xg, Ks[s]);
      lstm_scan<<<SCAN_WGS, 256, 0, stream>>>(xg, whhbf[s], hcat, hbuf,
                                              dir, dir * Hz, sync);
    }
  }
  fc_logits<<<Mz, 256, 0, stream>>>(hcat, fcw, fcb, out + 1);
  crf_loss<<<1, 128, 0, stream>>>(out + 1, labels, cstart, cend, ctrans, out);
}
